// AutoEncoder_1614907703387
// MI455X (gfx1250) — matrix operand order repaired and checked by reading, not compiled, not run on hardware
//
#include <hip/hip_runtime.h>
#include <math.h>

// ---------------------------------------------------------------------------
// GRU autoencoder, MI455X (gfx1250, wave32).
//   B=256, T=512, I=1, H=1024, CODE=3.
// Persistent kernel: 256 WGs x 128 threads, device-wide barrier per timestep.
// Recurrent GEMM h @ Whh.T runs on v_wmma_f32_16x16x32_bf16 with:
//   - A (h, bf16) streamed from L2 via global_load_b128 (double-buffered),
//   - B (Whh slice, bf16) resident in LDS (96 KB/WG, staged once per phase),
//   - f32 accumulators, f32 gate math on fast v_exp/v_rcp paths.
// ---------------------------------------------------------------------------

#define B_    256
#define T_    512
#define H_    1024
#define CODE_ 3
#define NWG   256
#define NTHR  128
#define HBANK ((size_t)B_ * H_)   // elements per h bank

typedef __attribute__((ext_vector_type(16))) __bf16 v16bf;
typedef __attribute__((ext_vector_type(8)))  __bf16 v8bf;
typedef __attribute__((ext_vector_type(8)))  float  v8f;

union ABf { v16bf v; v8bf h[2]; };

__device__ __forceinline__ float fast_rcp(float x) {
    return __builtin_amdgcn_rcpf(x);          // v_rcp_f32
}
__device__ __forceinline__ float sigmoidf_(float x) {
    return fast_rcp(1.0f + __expf(-x));       // v_exp_f32 + v_rcp_f32
}
__device__ __forceinline__ float tanhf_(float x) {
    float e = __expf(2.0f * x);
    return 1.0f - 2.0f * fast_rcp(e + 1.0f);  // exact at both saturations
}

// Device-wide barrier via monotonically increasing generation counter in ws.
// Replay-safe: each launch reads its starting generation before first use.
__device__ __forceinline__ void gridbar(unsigned* bar, volatile unsigned* gen,
                                        unsigned gen0, unsigned cnt) {
    __threadfence();
    __syncthreads();
    if (threadIdx.x == 0) {
        unsigned prev = atomicAdd(bar, 1u);
        if (prev == (unsigned)(NWG - 1)) {
            atomicExch(bar, 0u);           // reset for next round (all arrived)
            __threadfence();
            atomicAdd((unsigned*)gen, 1u); // release
        } else {
            while ((unsigned)(*gen - gen0) < cnt) {
                __builtin_amdgcn_s_sleep(2);
            }
        }
    }
    __syncthreads();
    __threadfence();
}

// One GRU-step GEMM for this wave's 16x16 tile: three gate accumulators,
// A from global bf16 h (read bank), B (3 gates) from LDS weight slice.
__device__ __forceinline__ void gemm3(const __bf16* __restrict__ hrd,
                                      const __bf16* __restrict__ sW,
                                      int m0, int lane,
                                      v8f& oR, v8f& oZ, v8f& oN) {
    v8f r = {}, z = {}, n = {};
    const int arow = m0 + (lane & 15);
    const int asel = (lane >> 4) ? 16 : 0;   // A: lanes16-31 hold K 16-31
    const int bsel = (lane >> 4) ? 16 : 0;   // B: lanes16-31 hold K 16-31
    const __bf16* ap = hrd + (size_t)arow * H_ + asel;
    const __bf16* bp = sW + (size_t)(lane & 15) * H_ + bsel;
#pragma unroll 2
    for (int k0 = 0; k0 < H_; k0 += 32) {
        ABf a, b0, b1, b2;
        a.h[0]  = *(const v8bf*)(ap + k0);
        a.h[1]  = *(const v8bf*)(ap + k0 + 8);
        __builtin_prefetch((const void*)(ap + k0 + 64), 0, 3);
        b0.h[0] = *(const v8bf*)(bp + k0);
        b0.h[1] = *(const v8bf*)(bp + k0 + 8);
        b1.h[0] = *(const v8bf*)(bp + 16 * H_ + k0);
        b1.h[1] = *(const v8bf*)(bp + 16 * H_ + k0 + 8);
        b2.h[0] = *(const v8bf*)(bp + 32 * H_ + k0);
        b2.h[1] = *(const v8bf*)(bp + 32 * H_ + k0 + 8);
        r = __builtin_amdgcn_wmma_f32_16x16x32_bf16(false, a.v, false, b0.v,
                                                    (short)0, r, false, false);
        z = __builtin_amdgcn_wmma_f32_16x16x32_bf16(false, a.v, false, b1.v,
                                                    (short)0, z, false, false);
        n = __builtin_amdgcn_wmma_f32_16x16x32_bf16(false, a.v, false, b2.v,
                                                    (short)0, n, false, false);
    }
    oR = r; oZ = z; oN = n;
}

extern "C" __global__ void __launch_bounds__(NTHR, 1)
gru_autoencoder(const float* __restrict__ in, const float* __restrict__ h0g,
                const float* __restrict__ WihE, const float* __restrict__ WhhE,
                const float* __restrict__ bihE, const float* __restrict__ bhhE,
                const float* __restrict__ Wenc, const float* __restrict__ benc,
                const float* __restrict__ Wdec, const float* __restrict__ bdec,
                const float* __restrict__ WihD, const float* __restrict__ WhhD,
                const float* __restrict__ bihD, const float* __restrict__ bhhD,
                const float* __restrict__ Wout, const float* __restrict__ bout,
                float* out, char* ws) {
    extern __shared__ char smem[];
    __bf16* sW   = (__bf16*)smem;                 // 96 KB weight slice (3 gates)
    float*  sX   = (float*)(smem + 96 * 1024);    // 256 f32: decoder feedback x
    float*  sRed = sX + B_;                       // 128 f32: loss reduction
    __shared__ unsigned sGen0;

    // Workspace layout. Bank selection is done with offsets on single base
    // pointers (kernarg-rooted GEPs) so loads/stores lower to global_*, not
    // flat_* (address-space inference survives).
    float*  hfB = (float*)ws;                      // 2 banks x 1 MB (f32 h)
    __bf16* hbB = (__bf16*)(ws + (2u << 20));      // 2 banks x 512 KB (bf16 h)
    float* part = (float*)(ws + (3u << 20));       // 2 banks x 64 x 256
    float* feat = part + 2 * 64 * B_;              // 256 x 3
    unsigned* bar = (unsigned*)(feat + B_ * CODE_);
    volatile unsigned* gen = bar + 1;

    const int tid  = threadIdx.x;
    const int lane = tid & 31;
    const int wv   = tid >> 5;                     // wave 0..3
    const int wg   = blockIdx.x;
    const int mblk = wg >> 6;                      // 0..3   (64 batch rows)
    const int nblk = wg & 63;                      // 0..63  (16 H-cols)
    const int m0   = mblk * 64 + wv * 16;
    const int c0   = nblk * 16;
    const int colL    = c0 + (lane & 15);          // this lane's H column
    const int rowBase = m0 + ((lane >> 4) << 3);   // C/D: lanes>=16 -> rows +8

    if (tid == 0) sGen0 = *gen;                    // replay-safe base generation
    __syncthreads();
    const unsigned gen0 = sGen0;
    unsigned barCnt = 0;

    // ---- init: h bank0 from hidden_state; stage encoder weights to LDS ----
    for (int i = 0; i < 8; ++i) {
        size_t o = (size_t)(rowBase + i) * H_ + colL;
        float v = h0g[o];
        hfB[o] = v;
        hbB[o] = (__bf16)v;
    }
    for (int idx = tid; idx < 48 * H_; idx += NTHR) {
        int rl = idx >> 10, k = idx & (H_ - 1);
        int g = rl >> 4, cl = rl & 15;
        sW[idx] = (__bf16)WhhE[(size_t)(g * H_ + c0 + cl) * H_ + k];
    }
    __syncthreads();
    gridbar(bar, gen, gen0, ++barCnt);

    // ---- encoder: 512 sequential GRU steps -------------------------------
    {
        const float wiR = WihE[colL], wiZ = WihE[H_ + colL], wiN = WihE[2 * H_ + colL];
        const float biR = bihE[colL], biZ = bihE[H_ + colL], biN = bihE[2 * H_ + colL];
        const float bhR = bhhE[colL], bhZ = bhhE[H_ + colL], bhN = bhhE[2 * H_ + colL];
        for (int t = 0; t < T_; ++t) {
            const size_t rb = (size_t)(t & 1) * HBANK;
            const size_t wb = (size_t)((t & 1) ^ 1) * HBANK;
            v8f aR, aZ, aN;
            gemm3(hbB + rb, sW, m0, lane, aR, aZ, aN);
#pragma unroll
            for (int i = 0; i < 8; ++i) {
                size_t o  = (size_t)(rowBase + i) * H_ + colL;
                float x  = in[(size_t)(rowBase + i) * T_ + t];
                float r  = sigmoidf_(aR[i] + bhR + x * wiR + biR);
                float z  = sigmoidf_(aZ[i] + bhZ + x * wiZ + biZ);
                float n  = tanhf_(x * wiN + biN + r * (aN[i] + bhN));
                float ho = hfB[rb + o];
                float hn = (1.0f - z) * n + z * ho;
                hfB[wb + o] = hn;
                hbB[wb + o] = (__bf16)hn;
            }
            gridbar(bar, gen, gen0, ++barCnt);
        }
    }

    // ---- bottleneck: features = sigmoid(h_T @ Wenc.T + benc)  (h_T: bank0)
    if (wg == 0) {
        for (int b = tid; b < B_; b += NTHR) {
            const float* hr = hfB + (size_t)b * H_;
            for (int j = 0; j < CODE_; ++j) {
                float s = benc[j];
                const float* wj = Wenc + (size_t)j * H_;
                for (int k = 0; k < H_; ++k) s += hr[k] * wj[k];
                float f = sigmoidf_(s);
                feat[b * CODE_ + j] = f;
                out[1 + b * CODE_ + j] = f;           // d_out: features
            }
        }
    }
    // restage decoder weights while the feature barrier is pending
    __syncthreads();
    for (int idx = tid; idx < 48 * H_; idx += NTHR) {
        int rl = idx >> 10, k = idx & (H_ - 1);
        int g = rl >> 4, cl = rl & 15;
        sW[idx] = (__bf16)WhhD[(size_t)(g * H_ + c0 + cl) * H_ + k];
    }
    __syncthreads();
    gridbar(bar, gen, gen0, ++barCnt);               // features visible

    // h0 = features @ Wdec.T + bdec  -> bank0
    for (int i = 0; i < 8; ++i) {
        size_t o = (size_t)(rowBase + i) * H_ + colL;
        float s = bdec[colL];
        for (int j = 0; j < CODE_; ++j)
            s += feat[(rowBase + i) * CODE_ + j] * Wdec[colL * CODE_ + j];
        hfB[o] = s;
        hbB[o] = (__bf16)s;
    }
    gridbar(bar, gen, gen0, ++barCnt);               // h0 visible everywhere

    // ---- decoder: 512 autoregressive GRU steps ---------------------------
    {
        const float wiR = WihD[colL], wiZ = WihD[H_ + colL], wiN = WihD[2 * H_ + colL];
        const float biR = bihD[colL], biZ = bihD[H_ + colL], biN = bihD[2 * H_ + colL];
        const float bhR = bhhD[colL], bhZ = bhhD[H_ + colL], bhN = bhhD[2 * H_ + colL];
        const float wo = Wout[colL];
        const float bo = bout[0];
        for (int t = 0; t < T_; ++t) {
            const size_t rb = (size_t)(t & 1) * HBANK;
            const size_t wb = (size_t)((t & 1) ^ 1) * HBANK;
            const int pbW = t & 1, pbR = pbW ^ 1;    // partial banks (write/read)
            if (t > 0) {
                // x_{t} = y_{t-1}: fixed-order reduction of per-column partials
                for (int r = tid; r < B_; r += NTHR) {
                    float s = bo;
                    const float* pp = part + pbR * 64 * B_ + r;
                    for (int nb = 0; nb < 64; ++nb) s += pp[nb * B_];
                    sX[r] = s;
                    if (wg == 0) out[1 + B_ * CODE_ + r * T_ + (t - 1)] = s;
                }
                __syncthreads();
            }
            v8f aR, aZ, aN;
            gemm3(hbB + rb, sW, m0, lane, aR, aZ, aN);
            float rs[8];
#pragma unroll
            for (int i = 0; i < 8; ++i) {
                size_t o  = (size_t)(rowBase + i) * H_ + colL;
                float x  = (t == 0) ? 0.0f : sX[rowBase + i];
                float r  = sigmoidf_(aR[i] + bhR + x * wiR + biR);
                float z  = sigmoidf_(aZ[i] + bhZ + x * wiZ + biZ);
                float n  = tanhf_(x * wiN + biN + r * (aN[i] + bhN));
                float ho = hfB[rb + o];
                float hn = (1.0f - z) * n + z * ho;
                hfB[wb + o] = hn;
                hbB[wb + o] = (__bf16)hn;
                rs[i] = hn * wo;                      // Wout partial, this column
            }
            // sum this tile's 16 columns per row (butterfly within 16 lanes)
            for (int m = 1; m < 16; m <<= 1)
#pragma unroll
                for (int i = 0; i < 8; ++i) rs[i] += __shfl_xor(rs[i], m, 16);
            if ((lane & 15) == 0) {
#pragma unroll
                for (int i = 0; i < 8; ++i)
                    part[pbW * 64 * B_ + nblk * B_ + (rowBase + i)] = rs[i];
            }
            gridbar(bar, gen, gen0, ++barCnt);
        }
    }

    // ---- epilogue (WG0): last output column, then MSE loss ---------------
    if (wg == 0) {
        const float bo = bout[0];
        const int pb = (T_ - 1) & 1;
        for (int r = tid; r < B_; r += NTHR) {
            float s = bo;
            const float* pp = part + pb * 64 * B_ + r;
            for (int nb = 0; nb < 64; ++nb) s += pp[nb * B_];
            out[1 + B_ * CODE_ + r * T_ + (T_ - 1)] = s;
        }
        __threadfence();
        __syncthreads();
        float acc = 0.0f;
        for (int idx = tid; idx < B_ * T_; idx += NTHR) {
            float d = in[idx] - out[1 + B_ * CODE_ + idx];
            acc += d * d;
        }
        sRed[tid] = acc;
        __syncthreads();
        for (int s2 = NTHR / 2; s2 > 0; s2 >>= 1) {
            if (tid < s2) sRed[tid] += sRed[tid + s2];
            __syncthreads();
        }
        if (tid == 0) out[0] = sRed[0] * (1.0f / (float)(B_ * T_));
    }
}

extern "C" void kernel_launch(void* const* d_in, const int* in_sizes, int n_in,
                              void* d_out, int out_size, void* d_ws, size_t ws_size,
                              hipStream_t stream) {
    (void)in_sizes; (void)n_in; (void)out_size; (void)ws_size;
    const float* in   = (const float*)d_in[0];
    const float* h0   = (const float*)d_in[1];
    const float* WihE = (const float*)d_in[2];
    const float* WhhE = (const float*)d_in[3];
    const float* bihE = (const float*)d_in[4];
    const float* bhhE = (const float*)d_in[5];
    const float* Wenc = (const float*)d_in[6];
    const float* benc = (const float*)d_in[7];
    const float* Wdec = (const float*)d_in[8];
    const float* bdec = (const float*)d_in[9];
    const float* WihD = (const float*)d_in[10];
    const float* WhhD = (const float*)d_in[11];
    const float* bihD = (const float*)d_in[12];
    const float* bhhD = (const float*)d_in[13];
    const float* Wout = (const float*)d_in[14];
    const float* bout = (const float*)d_in[15];

    const size_t shmem = 96 * 1024 + (B_ + NTHR) * sizeof(float);
    gru_autoencoder<<<dim3(NWG), dim3(NTHR), shmem, stream>>>(
        in, h0, WihE, WhhE, bihE, bhhE, Wenc, benc, Wdec, bdec,
        WihD, WhhD, bihD, bhhD, Wout, bout, (float*)d_out, (char*)d_ws);
}
